// GATAutoEncoder_4020089389721
// MI455X (gfx1250) — matrix operand order repaired and checked by reading, not compiled, not run on hardware
//
#include <hip/hip_runtime.h>
#include <hip/hip_bf16.h>

// ---------------------------------------------------------------------------
// GAT autoencoder forward for MI455X (gfx1250, wave32, WMMA).
// N=100000 nodes, E=1.6M edges, H=8 heads, O=16, FEAT=128, EMB=64.
// GEMMs: v_wmma_f32_16x16x32_f16 with pre-packed B fragments (L2-resident,
// contiguous b128 loads, zero LDS). Edge phase (roofline term) uses float4
// gathers + f32 atomics, with per-head attention broadcast via __shfl.
// ---------------------------------------------------------------------------

typedef __attribute__((ext_vector_type(16))) _Float16 v16h;
typedef __attribute__((ext_vector_type(8)))  float    v8f;

#define HEADS 8
#define FEAT  128   // H*O
#define NEG_SLOPE 0.2f

// ---------------- f32 -> f16 conversion (activations) ----------------
__global__ void cvt_f32_f16_kernel(const float* __restrict__ in,
                                   _Float16* __restrict__ out, int n) {
  int i = blockIdx.x * blockDim.x + threadIdx.x;
  if (i < n) out[i] = (_Float16)in[i];
}

// ---------------- pack weight B[K=128, Ncol] into WMMA fragment order ------
// Output layout: Bp[((nt*4 + kt)*32 + lane)*16 + i], where lane = hi*16 + c,
// k = kt*32 + hi*16 + i, col = nt*16 + c.  Each wave then reads its fragment
// as one contiguous 32-byte v16h.
__global__ void pack_B_frag_kernel(const float* __restrict__ B,
                                   _Float16* __restrict__ Bp, int Ncol) {
  int idx = blockIdx.x * blockDim.x + threadIdx.x;
  int total = 128 * Ncol;
  if (idx >= total) return;
  int i  = idx & 15;
  int l  = (idx >> 4) & 31;
  int kt = (idx >> 9) & 3;          // K/32 = 4
  int nt = idx >> 11;               // per-ntile block = 2048 halfs
  int c  = l & 15;
  int hi = l >> 4;
  int k  = kt * 32 + hi * 16 + i;
  int col = nt * 16 + c;
  Bp[idx] = (_Float16)B[(size_t)k * Ncol + col];
}

// ---------------- WMMA GEMM: C[M,Ncol] = A[M,128] * Bp (+bias) -------------
// A f16 row-major, Bp fragment-packed f16, C f32 row-major.
// 8 waves per 256-thread block, one 16x16 tile per wave, no LDS.
__global__ void wmma_gemm_k128(const _Float16* __restrict__ A,
                               const _Float16* __restrict__ Bp,
                               float* __restrict__ C,
                               const float* __restrict__ bias,
                               int M, int Ncol) {
  const int lane = threadIdx.x & 31;
  const int wave = threadIdx.x >> 5;            // 0..7
  const int ntx  = Ncol >> 4;                   // col tiles
  const int wid  = blockIdx.x * 8 + wave;
  const int col0 = (wid % ntx) << 4;
  const int row0 = (wid / ntx) << 4;
  if (row0 >= M) return;                        // wave-uniform

  const int mloc = lane & 15;
  const int hi   = lane >> 4;
  const _Float16* Arow  = A + (size_t)(row0 + mloc) * 128;
  const _Float16* Bfrag = Bp + ((size_t)(col0 >> 4) * 4 * 32 + lane) * 16;

  v8f acc = {};
  #pragma unroll
  for (int kt = 0; kt < 4; ++kt) {
    if (kt < 3) __builtin_prefetch(Arow + (kt + 1) * 32, 0, 0);
    v16h a, b;
    // A fragment: k = 16*hi + s (one contiguous 16-half run per lane, matching Bp).
    #pragma unroll
    for (int v = 0; v < 8; ++v) {
      int k = kt * 32 + (hi << 4) + (v << 1);
      a[2 * v]     = Arow[k];
      a[2 * v + 1] = Arow[k + 1];
    }
    // B fragment: one contiguous 32-byte load (two b128s) from L2.
    b = *(const v16h*)(Bfrag + (size_t)kt * 32 * 16);
    acc = __builtin_amdgcn_wmma_f32_16x16x32_f16(
        false, a, false, b, (short)0, acc, false, false);
  }

  float bs = bias ? bias[col0 + mloc] : 0.0f;
  float* Crow = C + (size_t)(row0 + (hi << 3)) * Ncol + col0 + mloc;
  #pragma unroll
  for (int v = 0; v < 8; ++v) Crow[(size_t)v * Ncol] = acc[v] + bs;
}

// ---------------- attention logits + self-loop max seed ----------------
__global__ void att_logits_kernel(const float* __restrict__ Alin,
                                  const float* __restrict__ a_src,
                                  const float* __restrict__ a_dst,
                                  float* __restrict__ als,
                                  float* __restrict__ ald,
                                  float* __restrict__ amax, int Nn) {
  int t = blockIdx.x * blockDim.x + threadIdx.x;
  if (t >= Nn * HEADS) return;
  int n = t >> 3, h = t & 7;
  const float* hv = Alin + (size_t)n * FEAT + h * 16;
  const float* as = a_src + h * 16;
  const float* ad = a_dst + h * 16;
  float s = 0.f, d = 0.f;
  #pragma unroll
  for (int o = 0; o < 16; ++o) { float x = hv[o]; s += x * as[o]; d += x * ad[o]; }
  als[t] = s; ald[t] = d;
  float a = s + d;                       // self-loop alpha
  amax[t] = (a > 0.f) ? a : a * NEG_SLOPE;
}

__device__ __forceinline__ void atomicMaxFloatOrdered(float* addr, float val) {
  int iv = __float_as_int(val);
  if (iv >= 0) atomicMax((int*)addr, iv);
  else         atomicMin((unsigned int*)addr, (unsigned int)iv);
}

__device__ __forceinline__ void load8(const float* __restrict__ p, float* r) {
  float4 a = ((const float4*)p)[0];
  float4 b = ((const float4*)p)[1];
  r[0] = a.x; r[1] = a.y; r[2] = a.z; r[3] = a.w;
  r[4] = b.x; r[5] = b.y; r[6] = b.z; r[7] = b.w;
}

// ---------------- edge pass 1: segment max ----------------
__global__ void edge_max_kernel(const int* __restrict__ src, const int* __restrict__ dst,
                                const float* __restrict__ als, const float* __restrict__ ald,
                                float* __restrict__ amax, int E) {
  int e = blockIdx.x * blockDim.x + threadIdx.x;
  if (e >= E) return;
  int s = src[e], d = dst[e];
  float as_[8], ad_[8];
  load8(als + (size_t)s * 8, as_);
  load8(ald + (size_t)d * 8, ad_);
  #pragma unroll
  for (int h = 0; h < HEADS; ++h) {
    float a = as_[h] + ad_[h];
    a = (a > 0.f) ? a : a * NEG_SLOPE;
    atomicMaxFloatOrdered(&amax[(size_t)d * 8 + h], a);
  }
}

// ---------------- denominator init with self-loop term ----------------
__global__ void den_init_kernel(const float* __restrict__ als, const float* __restrict__ ald,
                                const float* __restrict__ amax, float* __restrict__ den, int Nn) {
  int t = blockIdx.x * blockDim.x + threadIdx.x;
  if (t >= Nn * HEADS) return;
  float a = als[t] + ald[t];
  a = (a > 0.f) ? a : a * NEG_SLOPE;
  den[t] = __expf(a - amax[t]);
}

// ---------------- edge pass 2: segment sum of exp ----------------
__global__ void edge_den_kernel(const int* __restrict__ src, const int* __restrict__ dst,
                                const float* __restrict__ als, const float* __restrict__ ald,
                                const float* __restrict__ amax, float* __restrict__ den, int E) {
  int e = blockIdx.x * blockDim.x + threadIdx.x;
  if (e >= E) return;
  int s = src[e], d = dst[e];
  float as_[8], ad_[8], am_[8];
  load8(als + (size_t)s * 8, as_);
  load8(ald + (size_t)d * 8, ad_);
  load8(amax + (size_t)d * 8, am_);
  #pragma unroll
  for (int h = 0; h < HEADS; ++h) {
    float a = as_[h] + ad_[h];
    a = (a > 0.f) ? a : a * NEG_SLOPE;
    atomicAdd(&den[(size_t)d * 8 + h], __expf(a - am_[h]));
  }
}

// ---------------- self-loop contribution initializes out ----------------
__global__ void out_self_kernel(const float* __restrict__ Alin,
                                const float* __restrict__ als, const float* __restrict__ ald,
                                const float* __restrict__ amax, const float* __restrict__ den,
                                float* __restrict__ out, int Nn) {
  int t = blockIdx.x * blockDim.x + threadIdx.x;
  if (t >= Nn * HEADS) return;
  int n = t >> 3, h = t & 7;
  float a = als[t] + ald[t];
  a = (a > 0.f) ? a : a * NEG_SLOPE;
  float att = __expf(a - amax[t]) / (den[t] + 1e-16f);
  const float* hv = Alin + (size_t)n * FEAT + h * 16;
  float* ov = out + (size_t)n * FEAT + h * 16;
  #pragma unroll
  for (int o = 0; o < 16; ++o) ov[o] = att * hv[o];
}

// ---------------- edge pass 3: weighted scatter-add (wave per edge) -------
// Lanes 0..7 gather one head's attention params (coalesced 32B rows); the
// per-head attention weight is broadcast to the 4 lanes owning that head's
// 16 output columns via __shfl.
__global__ void edge_aggr_kernel(const int* __restrict__ src, const int* __restrict__ dst,
                                 const float* __restrict__ Alin,
                                 const float* __restrict__ als, const float* __restrict__ ald,
                                 const float* __restrict__ amax, const float* __restrict__ den,
                                 float* __restrict__ out, int E) {
  int lane = threadIdx.x & 31;
  int e = blockIdx.x * (blockDim.x >> 5) + (threadIdx.x >> 5);
  if (e >= E) return;
  int s = src[e], d = dst[e];
  float att = 0.f;
  if (lane < HEADS) {
    float a = als[(size_t)s * 8 + lane] + ald[(size_t)d * 8 + lane];
    a = (a > 0.f) ? a : a * NEG_SLOPE;
    att = __expf(a - amax[(size_t)d * 8 + lane]) / (den[(size_t)d * 8 + lane] + 1e-16f);
  }
  att = __shfl(att, lane >> 2, 32);      // broadcast head weight to its 4 lanes
  const float4 hv = *(const float4*)(Alin + (size_t)s * FEAT + lane * 4);
  float* ov = out + (size_t)d * FEAT + lane * 4;
  atomicAdd(ov + 0, att * hv.x);
  atomicAdd(ov + 1, att * hv.y);
  atomicAdd(ov + 2, att * hv.z);
  atomicAdd(ov + 3, att * hv.w);
}

// ---------------- bias + relu + f16 convert (fused) ----------------
__global__ void bias_relu_cvt_kernel(float* __restrict__ out, const float* __restrict__ bias,
                                     _Float16* __restrict__ hf16, int total) {
  int i = blockIdx.x * blockDim.x + threadIdx.x;
  if (i >= total) return;
  float v = out[i] + bias[i & (FEAT - 1)];
  v = (v > 0.f) ? v : 0.f;
  out[i] = v;
  hf16[i] = (_Float16)v;
}

// ---------------- decoder: z[64] -> 16 -> BN -> 64 -> 128 ----------------
__global__ void decoder_kernel(const float* __restrict__ z,
                               const float* __restrict__ Wd1, const float* __restrict__ bd1,
                               const float* __restrict__ bn_g, const float* __restrict__ bn_b,
                               const float* __restrict__ Wd2, const float* __restrict__ bd2,
                               const float* __restrict__ Wd3, const float* __restrict__ bd3,
                               float* __restrict__ xr, int Nn) {
  __shared__ float sW1[64 * 16], sW2[16 * 64], sW3[64 * 128];
  __shared__ float sb1[16], sg[16], sbb[16], sb2[64], sb3[128];
  for (int i = threadIdx.x; i < 64 * 16; i += blockDim.x) sW1[i] = Wd1[i];
  for (int i = threadIdx.x; i < 16 * 64; i += blockDim.x) sW2[i] = Wd2[i];
  for (int i = threadIdx.x; i < 64 * 128; i += blockDim.x) sW3[i] = Wd3[i];
  if (threadIdx.x < 16) {
    sb1[threadIdx.x] = bd1[threadIdx.x];
    sg[threadIdx.x]  = bn_g[threadIdx.x];
    sbb[threadIdx.x] = bn_b[threadIdx.x];
  }
  for (int i = threadIdx.x; i < 64;  i += blockDim.x) sb2[i] = bd2[i];
  for (int i = threadIdx.x; i < 128; i += blockDim.x) sb3[i] = bd3[i];
  __syncthreads();

  const float inv = rsqrtf(1.0f + 1e-5f);
  for (int n = blockIdx.x * blockDim.x + threadIdx.x; n < Nn;
       n += gridDim.x * blockDim.x) {
    float zv[64];
    const float* zp = z + (size_t)n * 64;
    #pragma unroll
    for (int i = 0; i < 64; ++i) zv[i] = zp[i];

    float d1[16];
    #pragma unroll
    for (int j = 0; j < 16; ++j) {
      float s = sb1[j];
      for (int i = 0; i < 64; ++i) s += zv[i] * sW1[i * 16 + j];
      s = (s > 0.f) ? s : 0.f;           // relu
      d1[j] = sg[j] * s * inv + sbb[j];  // eval BN
    }
    float d2[64];
    #pragma unroll
    for (int j = 0; j < 64; ++j) {
      float s = sb2[j];
      #pragma unroll
      for (int i = 0; i < 16; ++i) s += d1[i] * sW2[i * 64 + j];
      d2[j] = (s > 0.f) ? s : 0.f;
    }
    float* xo = xr + (size_t)n * 128;
    for (int j = 0; j < 128; ++j) {
      float s = sb3[j];
      for (int i = 0; i < 64; ++i) s += d2[i] * sW3[i * 128 + j];
      xo[j] = s;
    }
  }
}

// ---------------------------------------------------------------------------
static inline size_t align256(size_t x) { return (x + 255) & ~(size_t)255; }

extern "C" void kernel_launch(void* const* d_in, const int* in_sizes, int n_in,
                              void* d_out, int out_size, void* d_ws, size_t ws_size,
                              hipStream_t stream) {
  const float* x   = (const float*)d_in[0];
  const int*   ei  = (const int*)d_in[1];
  // d_in[2] = edge_weight (ignored by GATConv)
  const float* W1  = (const float*)d_in[3];
  const float* a1s = (const float*)d_in[4];
  const float* a1d = (const float*)d_in[5];
  const float* b1  = (const float*)d_in[6];
  const float* W2  = (const float*)d_in[7];
  const float* a2s = (const float*)d_in[8];
  const float* a2d = (const float*)d_in[9];
  const float* b2  = (const float*)d_in[10];
  const float* We  = (const float*)d_in[11];
  const float* be  = (const float*)d_in[12];
  const float* Wd1 = (const float*)d_in[13];
  const float* bd1 = (const float*)d_in[14];
  const float* bng = (const float*)d_in[15];
  const float* bnb = (const float*)d_in[16];
  const float* Wd2 = (const float*)d_in[17];
  const float* bd2 = (const float*)d_in[18];
  const float* Wd3 = (const float*)d_in[19];
  const float* bd3 = (const float*)d_in[20];

  const int N = in_sizes[0] / FEAT;      // 100000 (multiple of 16)
  const int E = in_sizes[1] / 2;         // 1600000
  const int* src = ei;
  const int* dst = ei + E;

  // workspace layout
  char* p = (char*)d_ws;
  _Float16* hf16 = (_Float16*)p; p += align256((size_t)N * FEAT * sizeof(_Float16));
  float* Alin    = (float*)p;    p += align256((size_t)N * FEAT * sizeof(float));
  float* outb    = (float*)p;    p += align256((size_t)N * FEAT * sizeof(float));
  float* als     = (float*)p;    p += align256((size_t)N * HEADS * sizeof(float));
  float* ald     = (float*)p;    p += align256((size_t)N * HEADS * sizeof(float));
  float* amax    = (float*)p;    p += align256((size_t)N * HEADS * sizeof(float));
  float* den     = (float*)p;    p += align256((size_t)N * HEADS * sizeof(float));
  _Float16* w1p  = (_Float16*)p; p += align256(128 * 128 * sizeof(_Float16));
  _Float16* w2p  = (_Float16*)p; p += align256(128 * 128 * sizeof(_Float16));
  _Float16* wep  = (_Float16*)p; p += align256(128 * 64 * sizeof(_Float16));

  float* z_out  = (float*)d_out + (size_t)N * FEAT;  // z section
  float* xr_out = (float*)d_out;                     // xr section

  const int TB = 256;
  const int nfeat = N * FEAT;
  const int nh = N * HEADS;
  dim3 blk(TB);

  // ---- precision staging: x -> f16, weights -> packed WMMA fragments ----
  cvt_f32_f16_kernel<<<dim3((nfeat + TB - 1) / TB), blk, 0, stream>>>(x, hf16, nfeat);
  pack_B_frag_kernel<<<dim3((128 * 128 + TB - 1) / TB), blk, 0, stream>>>(W1, w1p, 128);
  pack_B_frag_kernel<<<dim3((128 * 128 + TB - 1) / TB), blk, 0, stream>>>(W2, w2p, 128);
  pack_B_frag_kernel<<<dim3((128 * 64 + TB - 1) / TB), blk, 0, stream>>>(We, wep, 64);

  dim3 gemmBlk(TB);                                  // 8 waves per block
  const int tiles128 = (N / 16) * (FEAT / 16);
  const int tiles64  = (N / 16) * (64 / 16);
  dim3 gemmGrid128((tiles128 + 7) / 8);
  dim3 gemmGrid64((tiles64 + 7) / 8);
  dim3 nodeHG((nh + TB - 1) / TB);
  dim3 edgeG((E + TB - 1) / TB);
  dim3 aggrG((E + 7) / 8);                           // 8 waves (edges) per block

  // ================= GAT layer 1 =================
  wmma_gemm_k128<<<gemmGrid128, gemmBlk, 0, stream>>>(hf16, w1p, Alin, nullptr, N, FEAT);
  att_logits_kernel<<<nodeHG, blk, 0, stream>>>(Alin, a1s, a1d, als, ald, amax, N);
  edge_max_kernel<<<edgeG, blk, 0, stream>>>(src, dst, als, ald, amax, E);
  den_init_kernel<<<nodeHG, blk, 0, stream>>>(als, ald, amax, den, N);
  edge_den_kernel<<<edgeG, blk, 0, stream>>>(src, dst, als, ald, amax, den, E);
  out_self_kernel<<<nodeHG, blk, 0, stream>>>(Alin, als, ald, amax, den, outb, N);
  edge_aggr_kernel<<<aggrG, blk, 0, stream>>>(src, dst, Alin, als, ald, amax, den, outb, E);
  bias_relu_cvt_kernel<<<dim3((nfeat + TB - 1) / TB), blk, 0, stream>>>(outb, b1, hf16, nfeat);

  // ================= GAT layer 2 =================
  wmma_gemm_k128<<<gemmGrid128, gemmBlk, 0, stream>>>(hf16, w2p, Alin, nullptr, N, FEAT);
  att_logits_kernel<<<nodeHG, blk, 0, stream>>>(Alin, a2s, a2d, als, ald, amax, N);
  edge_max_kernel<<<edgeG, blk, 0, stream>>>(src, dst, als, ald, amax, E);
  den_init_kernel<<<nodeHG, blk, 0, stream>>>(als, ald, amax, den, N);
  edge_den_kernel<<<edgeG, blk, 0, stream>>>(src, dst, als, ald, amax, den, E);
  out_self_kernel<<<nodeHG, blk, 0, stream>>>(Alin, als, ald, amax, den, outb, N);
  edge_aggr_kernel<<<aggrG, blk, 0, stream>>>(src, dst, Alin, als, ald, amax, den, outb, E);
  bias_relu_cvt_kernel<<<dim3((nfeat + TB - 1) / TB), blk, 0, stream>>>(outb, b2, hf16, nfeat);

  // ================= embedding: z = h2 @ We + be (into d_out) =============
  wmma_gemm_k128<<<gemmGrid64, gemmBlk, 0, stream>>>(hf16, wep, z_out, be, N, 64);

  // ================= decoder =================
  decoder_kernel<<<dim3((N + TB - 1) / TB), blk, 0, stream>>>(
      z_out, Wd1, bd1, bng, bnb, Wd2, bd2, Wd3, bd3, xr_out, N);
}
